// Attn_spa_71863392797008
// MI455X (gfx1250) — compile-verified, matrix operand order checked by reading, not run on hardware
//
#include <hip/hip_runtime.h>
#include <hip/hip_bf16.h>

// ---------------------------------------------------------------------------
// Linformer attention for MI455X (gfx1250), wave32 + WMMA bf16.
//   x:(16,4096,512)  ->  Q=x@Wq ; KL=(pkT@x)@Wk ; VL=(pvT@x)@Wv
//   per (b,h): softmax(Q Kᵀ /8) V  ->  out@Wo+bo  (f32)
// All GEMMs use v_wmma_f32_16x16x32_bf16 with f32 accumulation.
// ---------------------------------------------------------------------------

typedef __attribute__((ext_vector_type(16))) __bf16 v16bf;
typedef __attribute__((ext_vector_type(8)))  float  v8f;

union Frag16 { v16bf v; int4 i[2]; };
union Acc    { v8f   v; float f[8]; };

__device__ __forceinline__ unsigned short f2bf(float x) {
    unsigned int u = __float_as_uint(x);
    u += 0x7FFFu + ((u >> 16) & 1u);          // round-to-nearest-even
    return (unsigned short)(u >> 16);
}

__device__ __forceinline__ v8f wmma_bf16(v16bf a, v16bf b, v8f c) {
    return __builtin_amdgcn_wmma_f32_16x16x32_bf16(
        /*neg_a=*/false, a, /*neg_b=*/false, b,
        /*c_mod=*/(short)0, c, /*reuse_a=*/false, /*reuse_b=*/false);
}

// A-style 16x32 bf16 fragment loader (ISA 7.12.2): lane L<16 row M=L holds
// K = {0..7, 16..23}; lane L+16 same row holds K = {8..15, 24..31}.
// Works for LDS (ds_load_b128) and global (global_load_b128) pointers.
__device__ __forceinline__ v16bf load_frag(const unsigned short* base,
                                           int row, int stride, int koff) {
    const int lane = (int)(threadIdx.x & 31u);
    const int hi   = (lane >> 4) & 1;
    const unsigned short* p = base + (long long)row * stride + koff + hi * 8;
    Frag16 f;
    f.i[0] = *(const int4*)(p);        // K 0..7   (or 8..15 in hi half)
    f.i[1] = *(const int4*)(p + 16);   // K 16..23 (or 24..31 in hi half)
    return f.v;
}

// ---------------------------------------------------------------------------
// dtype conversion helpers
// ---------------------------------------------------------------------------
__global__ void cvt_f32_bf16(const float* __restrict__ src,
                             unsigned short* __restrict__ dst, int count) {
    int i = blockIdx.x * blockDim.x + threadIdx.x;
    int stride = gridDim.x * blockDim.x;
    for (; i < count; i += stride) dst[i] = f2bf(src[i]);
}

// src: rows x cols (f32)  ->  dst: cols x rows (bf16)
__global__ void transpose_cvt(const float* __restrict__ src,
                              unsigned short* __restrict__ dst,
                              int rows, int cols) {
    int i = blockIdx.x * blockDim.x + threadIdx.x;
    int total = rows * cols;
    int stride = gridDim.x * blockDim.x;
    for (; i < total; i += stride) {
        int r = i / cols, c = i - r * cols;
        dst[(long long)c * rows + r] = f2bf(src[i]);
    }
}

// ---------------------------------------------------------------------------
// Generic bf16 WMMA GEMM:  C[M,N] = A[M,K] @ B[K,N]
// WG tile 128x128, BK=32, 8 waves (4x2), wave tile 32x64 (2x4 WMMA tiles).
// A staged in LDS row-major, B staged transposed so both use load_frag.
// STORE_BF16: bf16 store; else f32 store + optional bias.
// ---------------------------------------------------------------------------
template <bool STORE_BF16>
__global__ __launch_bounds__(256)
void gemm_bf16_kernel(const unsigned short* __restrict__ A,
                      const unsigned short* __restrict__ B,
                      void* __restrict__ Cv,
                      const float* __restrict__ bias,
                      int M, int N, int K, int lda, int ldb, int ldc,
                      long long sA, long long sB, long long sC) {
    __shared__ __align__(16) unsigned short As[128 * 32];
    __shared__ __align__(16) unsigned short Bs[128 * 32];   // [n][k]

    const int tid  = threadIdx.x;
    const int lane = tid & 31;
    const int w    = tid >> 5;       // 0..7
    const int wm   = w >> 1;         // 0..3  (M dir)
    const int wn   = w & 1;          // 0..1  (N dir)
    const int nl   = lane & 15;
    const int hi   = (lane >> 4) & 1;

    const int bz = blockIdx.z;
    const unsigned short* Ab = A + (long long)bz * sA;
    const unsigned short* Bb = B + (long long)bz * sB;

    const int m0 = blockIdx.y * 128;
    const int n0 = blockIdx.x * 128;

    Acc acc[2][4];
#pragma unroll
    for (int i = 0; i < 2; i++)
#pragma unroll
        for (int j = 0; j < 4; j++)
#pragma unroll
            for (int r = 0; r < 8; r++) acc[i][j].f[r] = 0.0f;

    for (int kk = 0; kk < K; kk += 32) {
        // ---- stage A tile (128x32), 16 bf16 per thread, b128 loads ----
        {
            int row = tid >> 1;
            int col = (tid & 1) * 16;
            const unsigned short* g = Ab + (long long)(m0 + row) * lda + kk + col;
            int4 v0 = *(const int4*)(g);
            int4 v1 = *(const int4*)(g + 8);
            *(int4*)(&As[row * 32 + col])     = v0;
            *(int4*)(&As[row * 32 + col + 8]) = v1;
        }
        // ---- stage B tile transposed: Bs[n][k] ----
#pragma unroll
        for (int c = 0; c < 2; c++) {
            int ch = tid + c * 256;          // 0..511 chunks of 8
            int kb = ch >> 4;                // 0..31
            int n  = (ch & 15) * 8;          // 0..120
            const unsigned short* g = Bb + (long long)(kk + kb) * ldb + n0 + n;
            int4 v = *(const int4*)(g);
            const unsigned short* pv = (const unsigned short*)&v;
#pragma unroll
            for (int i = 0; i < 8; i++) Bs[(n + i) * 32 + kb] = pv[i];
        }
        if (kk + 32 < K) {  // speculative prefetch of next K-slab
            __builtin_prefetch(Ab + (long long)(m0 + (tid >> 1)) * lda + kk + 32, 0, 1);
            __builtin_prefetch(Bb + (long long)(kk + 32 + (tid & 31)) * ldb + n0, 0, 1);
        }
        __syncthreads();

        v16bf af[2], bf[4];
#pragma unroll
        for (int i = 0; i < 2; i++)
            af[i] = load_frag(As, wm * 32 + i * 16 + nl, 32, 0);
#pragma unroll
        for (int j = 0; j < 4; j++)
            bf[j] = load_frag(Bs, wn * 64 + j * 16 + nl, 32, 0);
#pragma unroll
        for (int i = 0; i < 2; i++)
#pragma unroll
            for (int j = 0; j < 4; j++)
                acc[i][j].v = wmma_bf16(af[i], bf[j], acc[i][j].v);
        __syncthreads();
    }

    // ---- epilogue: C layout lane=N(0..15 per half), vgpr r => M=r(+8 hi) ----
#pragma unroll
    for (int i = 0; i < 2; i++)
#pragma unroll
        for (int j = 0; j < 4; j++)
#pragma unroll
            for (int r = 0; r < 8; r++) {
                int m = m0 + wm * 32 + i * 16 + r + hi * 8;
                int n = n0 + wn * 64 + j * 16 + nl;
                if (m < M && n < N) {
                    float v = acc[i][j].f[r];
                    if (STORE_BF16) {
                        unsigned short* C = (unsigned short*)Cv + (long long)bz * sC;
                        C[(long long)m * ldc + n] = f2bf(v);
                    } else {
                        float* C = (float*)Cv + (long long)bz * sC;
                        C[(long long)m * ldc + n] = v + (bias ? bias[n] : 0.0f);
                    }
                }
            }
}

// ---------------------------------------------------------------------------
// Fused Linformer attention: grid (n/128, heads, batch), 8 waves, 16 Q-rows
// per wave. K_low (256x64) + V_lowᵀ (64x256) live in LDS; scores + softmax
// stay in registers; probs repacked via wave-private LDS scratch for the
// second WMMA pass. Dynamic LDS = 32K + 32K + 8K = 72 KB.
// ---------------------------------------------------------------------------
__global__ __launch_bounds__(256)
void linformer_attn_kernel(const unsigned short* __restrict__ Q,
                           const unsigned short* __restrict__ KL,
                           const unsigned short* __restrict__ VL,
                           unsigned short* __restrict__ Out) {
    extern __shared__ __align__(16) unsigned short smem[];
    unsigned short* Ks = smem;                    // [256][64]
    unsigned short* Vt = smem + 256 * 64;         // [64][256]
    unsigned short* Ps = smem + 2 * 256 * 64;     // 8 waves * [16][32]

    const int tid  = threadIdx.x;
    const int lane = tid & 31;
    const int w    = tid >> 5;
    const int nl   = lane & 15;
    const int hi   = (lane >> 4) & 1;
    const int h    = blockIdx.y;
    const int b    = blockIdx.z;

    // ---- stage K_low (row-major) and V_lowᵀ into LDS ----
    const unsigned short* Kg = KL + (long long)b * 256 * 512 + h * 64;
    const unsigned short* Vg = VL + (long long)b * 256 * 512 + h * 64;
#pragma unroll
    for (int c = 0; c < 8; c++) {
        int ch  = tid + c * 256;         // 2048 chunks of 8
        int key = ch >> 3;               // 0..255
        int col = (ch & 7) * 8;          // 0..56
        int4 kv = *(const int4*)(Kg + (long long)key * 512 + col);
        *(int4*)(&Ks[key * 64 + col]) = kv;
        int4 vv = *(const int4*)(Vg + (long long)key * 512 + col);
        const unsigned short* pv = (const unsigned short*)&vv;
#pragma unroll
        for (int i = 0; i < 8; i++) Vt[(col + i) * 256 + key] = pv[i];
    }
    __syncthreads();

    // ---- Q fragments straight from global (d_h = 64 -> 2 K-steps) ----
    const long long qrow0 = (long long)b * 4096 + blockIdx.x * 128 + w * 16;
    const unsigned short* Qg = Q + (qrow0 + nl) * 512 + h * 64 + hi * 8;
    Frag16 qa[2];
#pragma unroll
    for (int s = 0; s < 2; s++) {
        qa[s].i[0] = *(const int4*)(Qg + s * 32);
        qa[s].i[1] = *(const int4*)(Qg + s * 32 + 16);
    }

    // ---- scores = Q @ K_lowᵀ : 16 tiles along the 256-key axis ----
    Acc sAcc[16];
#pragma unroll
    for (int t = 0; t < 16; t++)
#pragma unroll
        for (int r = 0; r < 8; r++) sAcc[t].f[r] = 0.0f;
#pragma unroll
    for (int t = 0; t < 16; t++) {
        v16bf k0 = load_frag(Ks, t * 16 + nl, 64, 0);
        v16bf k1 = load_frag(Ks, t * 16 + nl, 64, 32);
        sAcc[t].v = wmma_bf16(qa[0].v, k0, sAcc[t].v);
        sAcc[t].v = wmma_bf16(qa[1].v, k1, sAcc[t].v);
    }

    // ---- softmax over 256 keys. Row m = r + 8*hi lives in vgpr r across the
    //      16 lanes of one half-wave -> xor-shuffles 1/2/4/8 stay in-half. ----
    const float scale = 0.125f;   // 1/sqrt(64)
#pragma unroll
    for (int r = 0; r < 8; r++) {
        float mx = -3.4e38f;
#pragma unroll
        for (int t = 0; t < 16; t++) {
            float v = sAcc[t].f[r] * scale;
            sAcc[t].f[r] = v;
            mx = fmaxf(mx, v);
        }
        mx = fmaxf(mx, __shfl_xor(mx, 1, 32));
        mx = fmaxf(mx, __shfl_xor(mx, 2, 32));
        mx = fmaxf(mx, __shfl_xor(mx, 4, 32));
        mx = fmaxf(mx, __shfl_xor(mx, 8, 32));
        float sum = 0.0f;
#pragma unroll
        for (int t = 0; t < 16; t++) {
            float e = __expf(sAcc[t].f[r] - mx);
            sAcc[t].f[r] = e;
            sum += e;
        }
        sum += __shfl_xor(sum, 1, 32);
        sum += __shfl_xor(sum, 2, 32);
        sum += __shfl_xor(sum, 4, 32);
        sum += __shfl_xor(sum, 8, 32);
        float inv = 1.0f / sum;
#pragma unroll
        for (int t = 0; t < 16; t++) sAcc[t].f[r] *= inv;
    }

    // ---- out = probs @ V_low : repack probs (C-layout) -> A-fragment layout
    //      through a wave-private LDS scratch, 32 keys per step. ----
    unsigned short* myP = Ps + w * (16 * 32);
    Acc o[4];
#pragma unroll
    for (int j = 0; j < 4; j++)
#pragma unroll
        for (int r = 0; r < 8; r++) o[j].f[r] = 0.0f;

#pragma unroll
    for (int kb = 0; kb < 8; kb++) {
#pragma unroll
        for (int tt = 0; tt < 2; tt++) {
            int t = kb * 2 + tt;
#pragma unroll
            for (int r = 0; r < 8; r++)
                myP[(r + hi * 8) * 32 + tt * 16 + nl] = f2bf(sAcc[t].f[r]);
        }
        // same-wave LDS ops are in-order; compiler inserts s_wait_dscnt
        v16bf pa = load_frag(myP, nl, 32, 0);
#pragma unroll
        for (int j = 0; j < 4; j++) {
            v16bf vb = load_frag(Vt, j * 16 + nl, 256, kb * 32);
            o[j].v = wmma_bf16(pa, vb, o[j].v);
        }
    }

    // ---- store attention output (bf16, head-interleaved (b*n, 512)) ----
#pragma unroll
    for (int j = 0; j < 4; j++)
#pragma unroll
        for (int r = 0; r < 8; r++) {
            long long m = qrow0 + r + hi * 8;
            int n = h * 64 + j * 16 + nl;
            Out[m * 512 + n] = f2bf(o[j].f[r]);
        }
}

// ---------------------------------------------------------------------------
// host-side orchestration
// ---------------------------------------------------------------------------
extern "C" void kernel_launch(void* const* d_in, const int* in_sizes, int n_in,
                              void* d_out, int out_size, void* d_ws, size_t ws_size,
                              hipStream_t stream) {
    const float* x     = (const float*)d_in[0];   // (16,4096,512)
    const float* Wq    = (const float*)d_in[1];   // (512,512)
    const float* Wk    = (const float*)d_in[2];
    const float* Wv    = (const float*)d_in[3];
    const float* projk = (const float*)d_in[4];   // (4096,256)
    const float* projv = (const float*)d_in[5];
    const float* Wo    = (const float*)d_in[6];
    const float* bo    = (const float*)d_in[7];   // (512,)
    float* out = (float*)d_out;                   // (16,4096,512) f32

    char* ws = (char*)d_ws;
    size_t off = 0;
    auto alloc = [&](size_t bytes) -> void* {
        void* p = ws + off;
        off += (bytes + 255) & ~(size_t)255;
        return p;
    };
    const long long BN = 16ll * 4096;             // 65536 rows
    unsigned short* xb  = (unsigned short*)alloc(BN * 512 * 2);        // x bf16
    unsigned short* Qb  = (unsigned short*)alloc(BN * 512 * 2);        // Q bf16
    unsigned short* ao  = (unsigned short*)alloc(BN * 512 * 2);        // attn out
    unsigned short* wqb = (unsigned short*)alloc(512 * 512 * 2);
    unsigned short* wkb = (unsigned short*)alloc(512 * 512 * 2);
    unsigned short* wvb = (unsigned short*)alloc(512 * 512 * 2);
    unsigned short* wob = (unsigned short*)alloc(512 * 512 * 2);
    unsigned short* pkT = (unsigned short*)alloc(256ll * 4096 * 2);    // projkᵀ
    unsigned short* pvT = (unsigned short*)alloc(256ll * 4096 * 2);
    unsigned short* xkl = (unsigned short*)alloc(16ll * 256 * 512 * 2);
    unsigned short* xvl = (unsigned short*)alloc(16ll * 256 * 512 * 2);
    unsigned short* Klo = (unsigned short*)alloc(16ll * 256 * 512 * 2);
    unsigned short* Vlo = (unsigned short*)alloc(16ll * 256 * 512 * 2);

    dim3 blk(256);

    // dtype staging
    cvt_f32_bf16<<<2048, 256, 0, stream>>>(x, xb, (int)(BN * 512));
    cvt_f32_bf16<<<256, 256, 0, stream>>>(Wq, wqb, 512 * 512);
    cvt_f32_bf16<<<256, 256, 0, stream>>>(Wk, wkb, 512 * 512);
    cvt_f32_bf16<<<256, 256, 0, stream>>>(Wv, wvb, 512 * 512);
    cvt_f32_bf16<<<256, 256, 0, stream>>>(Wo, wob, 512 * 512);
    transpose_cvt<<<1024, 256, 0, stream>>>(projk, pkT, 4096, 256);
    transpose_cvt<<<1024, 256, 0, stream>>>(projv, pvT, 4096, 256);

    // Q = x @ Wq                               (65536x512x512)
    gemm_bf16_kernel<true><<<dim3(4, 512, 1), blk, 0, stream>>>(
        xb, wqb, Qb, nullptr, 65536, 512, 512, 512, 512, 512, 0, 0, 0);

    // xk_low[b] = projkᵀ @ x[b]                (256x512x4096, batched 16)
    gemm_bf16_kernel<true><<<dim3(4, 2, 16), blk, 0, stream>>>(
        pkT, xb, xkl, nullptr, 256, 512, 4096, 4096, 512, 512,
        0, 4096ll * 512, 256ll * 512);
    gemm_bf16_kernel<true><<<dim3(4, 2, 16), blk, 0, stream>>>(
        pvT, xb, xvl, nullptr, 256, 512, 4096, 4096, 512, 512,
        0, 4096ll * 512, 256ll * 512);

    // K_low = xk_low @ Wk ; V_low = xv_low @ Wv   (4096x512x512 flat)
    gemm_bf16_kernel<true><<<dim3(4, 32, 1), blk, 0, stream>>>(
        xkl, wkb, Klo, nullptr, 4096, 512, 512, 512, 512, 512, 0, 0, 0);
    gemm_bf16_kernel<true><<<dim3(4, 32, 1), blk, 0, stream>>>(
        xvl, wvb, Vlo, nullptr, 4096, 512, 512, 512, 512, 512, 0, 0, 0);

    // fused attention (softmax never touches HBM): 72 KB dynamic LDS
    linformer_attn_kernel<<<dim3(32, 8, 16), blk, 73728, stream>>>(Qb, Klo, Vlo, ao);

    // out = attn_out @ Wo + bo  (f32 store)
    gemm_bf16_kernel<false><<<dim3(4, 512, 1), blk, 0, stream>>>(
        ao, wob, out, bo, 65536, 512, 512, 512, 512, 512, 0, 0, 0);
}
